// L2P_54563264528725
// MI455X (gfx1250) — matrix operand order repaired and checked by reading, not compiled, not run on hardware
//
#include <hip/hip_runtime.h>

typedef __attribute__((ext_vector_type(16))) _Float16 v16h;
typedef __attribute__((ext_vector_type(8)))  float    v8f;
typedef __attribute__((ext_vector_type(4)))  float    v4f;

#define NB    8192   // batch
#define DKDIM 768    // key dim
#define DDIM  768    // embed dim
#define POOL  30
#define PLEN  20
#define TOPK  5

// ---------------------------------------------------------------------------
// Kernel A: cosine-sim top-5 indices via WMMA.
//   - K rows L2-normalized to f16 in LDS (rows 30,31 zero padded to N=32).
//   - Query rows left unnormalized (positive row scale preserves ranking).
//   - Each wave computes one 16-row x 32-col sim tile with
//     v_wmma_f32_16x16x32_f16, then a 5-pass argmax per row.
// ---------------------------------------------------------------------------
__global__ __launch_bounds__(256) void l2p_topk_kernel(
    const float* __restrict__ xq,   // [NB, DKDIM]
    const float* __restrict__ K,    // [POOL, DKDIM]
    int* __restrict__ idx_out)      // [NB, TOPK]
{
    __shared__ __align__(16) unsigned char smem_raw[32 * DKDIM * 2];  // 48 KB
    _Float16* nk = (_Float16*)smem_raw;

    const int tid  = threadIdx.x;
    const int lane = tid & 31;
    const int wv   = tid >> 5;          // wave in block: 0..7

    // ---- normalize K rows -> f16 LDS; rows >= POOL zero-filled ----
    for (int i = 0; i < 4; ++i) {
        const int row = wv + 8 * i;     // 0..31 (uniform per wave)
        if (row < POOL) {
            const float* kr = K + row * DKDIM;
            float s = 0.f;
            for (int c = lane; c < DKDIM; c += 32) { float v = kr[c]; s += v * v; }
            #pragma unroll
            for (int off = 16; off >= 1; off >>= 1) s += __shfl_xor(s, off, 32);
            const float rn = rsqrtf(fmaxf(s, 1e-24f));   // norm clamp 1e-12
            for (int c = lane; c < DKDIM; c += 32)
                nk[row * DKDIM + c] = (_Float16)(kr[c] * rn);
        } else {
            for (int c = lane; c < DKDIM; c += 32)
                nk[row * DKDIM + c] = (_Float16)0.f;
        }
    }
    __syncthreads();

    // ---- WMMA tile: 16 query rows x 32 pool cols, K = 768 ----
    const int waveGlobal = blockIdx.x * 8 + wv;
    const int m0  = waveGlobal * 16;
    const int hi  = (lane >> 4) & 1;           // which half-wave
    const int ln  = lane & 15;
    const float*    qrow = xq + (size_t)(m0 + ln) * DKDIM;
    const _Float16* bc0  = nk + (size_t)ln * DKDIM + hi * 16;          // cols 0..15
    const _Float16* bc1  = nk + (size_t)(ln + 16) * DKDIM + hi * 16;   // cols 16..31

    v8f c0 = {}; v8f c1 = {};
    for (int kb = 0; kb < DKDIM; kb += 32) {
        // A fragment (contiguous layout): lane holds K = kb + hi*16 + {0..15}
        const float* q1 = qrow + kb + hi * 16;
        v4f qa = *(const v4f*)(q1);
        v4f qb = *(const v4f*)(q1 + 4);
        v4f qc = *(const v4f*)(q1 + 8);
        v4f qd = *(const v4f*)(q1 + 12);
        v16h a;
        #pragma unroll
        for (int j = 0; j < 4; ++j) {
            a[j]      = (_Float16)qa[j];
            a[4 + j]  = (_Float16)qb[j];
            a[8 + j]  = (_Float16)qc[j];
            a[12 + j] = (_Float16)qd[j];
        }
        // B fragments: contiguous 16 halves of the (column) pool row
        v16h b0 = *(const v16h*)(bc0 + kb);
        v16h b1 = *(const v16h*)(bc1 + kb);
        c0 = __builtin_amdgcn_wmma_f32_16x16x32_f16(false, a, false, b0,
                                                    (short)0, c0, false, false);
        c1 = __builtin_amdgcn_wmma_f32_16x16x32_f16(false, a, false, b1,
                                                    (short)0, c1, false, false);
    }

    __syncthreads();                       // everyone done with nk; reuse LDS
    float* simbuf = (float*)smem_raw;      // [8 waves][16 rows][32 cols] = 16 KB
    float* mysim  = simbuf + wv * 512;
    #pragma unroll
    for (int r = 0; r < 8; ++r) {          // C/D layout: VGPR r -> M = r + hi*8
        const int M = r + hi * 8;
        mysim[M * 32 + ln]      = c0[r];
        mysim[M * 32 + 16 + ln] = c1[r];
    }
    __syncthreads();

    // ---- top-5 per row (strict > : lowest index wins ties, matches lax.top_k)
    if (lane < 16) {
        const float* v  = mysim + lane * 32;
        const int grow  = m0 + lane;
        unsigned chosen = 0u;
        #pragma unroll
        for (int j = 0; j < TOPK; ++j) {
            float best = -3.4e38f; int bi = 0;
            #pragma unroll
            for (int n = 0; n < POOL; ++n) {
                const float s = v[n];
                if (!((chosen >> n) & 1u) && s > best) { best = s; bi = n; }
            }
            chosen |= (1u << bi);
            idx_out[grow * TOPK + j] = bi;
        }
    }
}

// ---------------------------------------------------------------------------
// Kernel B: the bandwidth kernel. 2.5 GB of gathered prompt writes.
//   One 192-thread block per output row of 768 floats; each lane moves one
//   16-byte vector. p stays L2-hot; output stored nontemporal (NT b128)
//   to spare the 192 MB L2 from 2.5 GB of write-once data.
//   Output layout: [2][NB][TOPK*10][DDIM]; block id = b*100 + k*20 + t.
// ---------------------------------------------------------------------------
__global__ __launch_bounds__(192) void l2p_gather_kernel(
    const float* __restrict__ p,     // [POOL, PLEN, DDIM]
    const int*   __restrict__ idx,   // [NB, TOPK]
    float*       __restrict__ out)   // [2, NB, 50, DDIM]
{
    const int blk = blockIdx.x;            // b*100 + k*20 + t
    const int b   = blk / 100;
    const int rem = blk - b * 100;
    const int k   = rem / 20;
    const int t   = rem - k * 20;

    const int pool = idx[b * TOPK + k];
    const v4f* src = (const v4f*)(p + ((size_t)pool * PLEN + t) * DDIM);

    const int half = (t >= PLEN / 2) ? 1 : 0;
    const int tt   = t - half * (PLEN / 2);
    const size_t outRow = (size_t)half * (size_t)NB * (TOPK * PLEN / 2)
                        + (size_t)b * (TOPK * PLEN / 2)
                        + (size_t)(k * (PLEN / 2) + tt);
    v4f* dst = (v4f*)(out + outRow * (size_t)DDIM);

    const int c = threadIdx.x;             // 0..191 (768 floats / 4)
    v4f val = src[c];
    __builtin_nontemporal_store(val, dst + c);
}

// ---------------------------------------------------------------------------
extern "C" void kernel_launch(void* const* d_in, const int* in_sizes, int n_in,
                              void* d_out, int out_size, void* d_ws, size_t ws_size,
                              hipStream_t stream)
{
    const float* xq = (const float*)d_in[0];   // x_query [8192,768]
    // d_in[1] = x (pass-through in reference; unused in output)
    const float* K  = (const float*)d_in[2];   // [30,768]
    const float* p  = (const float*)d_in[3];   // [30,20,768]
    // d_in[4] = layer_id (branch always taken for layer 0)

    int*   idx = (int*)d_ws;                   // 8192*5 ints = 160 KB scratch
    float* out = (float*)d_out;                // [2,8192,50,768]

    // 512 waves total: 64 blocks x 8 waves, each wave owns a 16-row tile.
    l2p_topk_kernel<<<dim3(NB / (16 * 8)), dim3(256), 0, stream>>>(xq, K, idx);

    // One block per output row: 8192 * 5 * 20 = 819200 blocks.
    l2p_gather_kernel<<<dim3(NB * TOPK * PLEN), dim3(192), 0, stream>>>(p, idx, out);
}
